// PhysicsGraphTransformer_38010460570134
// MI455X (gfx1250) — compile-verified, matrix operand order checked by reading, not run on hardware
//
#include <hip/hip_runtime.h>
#include <hip/hip_bf16.h>
#include <math.h>

typedef __attribute__((ext_vector_type(16))) __bf16 v16bf;
typedef __attribute__((ext_vector_type(8)))  __bf16 v8bf;
typedef __attribute__((ext_vector_type(8)))  float  v8f;

// ---------- helpers ----------
static __device__ __forceinline__ unsigned short f2bfb(float f) {
  unsigned u = __float_as_uint(f);
  unsigned r = u + 0x7FFFu + ((u >> 16) & 1u);   // round-to-nearest-even
  return (unsigned short)(r >> 16);
}

// order-preserving float <-> uint for atomicMax-based segment max
static __device__ __forceinline__ unsigned f32_enc(float f) {
  unsigned u = __float_as_uint(f);
  return (u & 0x80000000u) ? ~u : (u | 0x80000000u);
}
static __device__ __forceinline__ float f32_dec(unsigned u) {
  unsigned v = (u & 0x80000000u) ? (u & 0x7FFFFFFFu) : ~u;
  return __uint_as_float(v);
}

// ---------------------------------------------------------------------------
// elementwise f32 -> packed bf16 (8 elements / thread); n must be multiple of 8
// ---------------------------------------------------------------------------
__global__ __launch_bounds__(256) void cvt_bf16_kernel(
    const float* __restrict__ src, __bf16* __restrict__ dst, int n8)
{
  int i = blockIdx.x * 256 + threadIdx.x;
  if (i >= n8) return;
  const float4* p = (const float4*)src + (size_t)i * 2;
  float4 a = p[0], b = p[1];
  uint4 o;
  o.x = (unsigned)f2bfb(a.x) | ((unsigned)f2bfb(a.y) << 16);
  o.y = (unsigned)f2bfb(a.z) | ((unsigned)f2bfb(a.w) << 16);
  o.z = (unsigned)f2bfb(b.x) | ((unsigned)f2bfb(b.y) << 16);
  o.w = (unsigned)f2bfb(b.z) | ((unsigned)f2bfb(b.w) << 16);
  ((uint4*)dst)[i] = o;
}

// ---------------------------------------------------------------------------
// WMMA GEMM:  out[M,256] = A[M,K] @ W[256,K]^T + bias   (bf16 inputs, f32 acc)
//   GATHER==1: A row e is concat(hbf[src[e]], hbf[dst[e]])  (K==512)
//   EPI==0: store tile; EPI==1: ReLU+dot(w2[256]) -> atomicAdd out[M]
//   EPI==2: ReLU+dot(w2[2,256]) -> atomicAdd out[M,2]
// Block: 256 thr = 8 waves; wave w: rows [m0+16w,+16) x cols [n0,+64).
// B panel (64 rows x K bf16) staged in LDS via async global->LDS copies.
// ---------------------------------------------------------------------------
template<int K, int GATHER, int EPI>
__global__ __launch_bounds__(256) void gemm_wmma_kernel(
    const __bf16* __restrict__ Abf,    // dense A [M,K] (GATHER==0)
    const __bf16* __restrict__ Hbf,    // node features [*,256] (GATHER==1)
    const long long* __restrict__ srcI,
    const long long* __restrict__ dstI,
    const __bf16* __restrict__ Wbf,    // [256,K] bf16
    const float* __restrict__ bias,    // [256]
    const float* __restrict__ w2,      // EPI1: [256], EPI2: [2,256]
    float* __restrict__ out,
    int M, int Nout)
{
  constexpr int KP = K + 8;            // +16B row pad to spread LDS banks
  __shared__ __bf16 Bs[64 * KP];

  const int tid  = threadIdx.x;
  const int lane = tid & 31;
  const int wave = tid >> 5;
  const int mr   = lane & 15;
  const int hsel = lane >> 4;
  const int m0 = blockIdx.y * 128 + wave * 16;
  const int n0 = blockIdx.x * 64;

  // ---- stage B panel: rows n0..n0+63 of Wbf (contiguous 64*K slab) ----
  {
    const int row = tid >> 2;          // 0..63
    const int sub = tid & 3;
    const __bf16* grow = Wbf + (size_t)(n0 + row) * K;
    __bf16* lrow = &Bs[row * KP];
#pragma unroll
    for (int c16 = sub; c16 < K / 8; c16 += 4) {
      unsigned loff = (unsigned)(uintptr_t)(&lrow[c16 * 8]);
      const __bf16* g = grow + c16 * 8;
      asm volatile("global_load_async_to_lds_b128 %0, %1, off"
                   :: "v"(loff), "v"(g) : "memory");
    }
    asm volatile("s_wait_asynccnt 0x0" ::: "memory");
  }
  __syncthreads();

  // ---- A row pointers (clamped; invalid rows only feed unstored C rows) ----
  const int mA = m0 + mr;
  const bool mvalid = (mA < M);
  const __bf16* arow0;
  const __bf16* arow1;
  if (GATHER) {
    long long gs = mvalid ? srcI[mA] : 0;
    long long gd = mvalid ? dstI[mA] : 0;
    arow0 = Hbf + gs * 256;            // serves K in [0,256)
    arow1 = Hbf + gd * 256 - 256;      // serves K in [256,512)
  } else {
    arow0 = Abf + (long long)(mvalid ? mA : 0) * K;
    arow1 = arow0;
  }

  const __bf16* brow[4];
#pragma unroll
  for (int t = 0; t < 4; ++t) brow[t] = &Bs[(t * 16 + mr) * KP];

  v8f zero;
#pragma unroll
  for (int i = 0; i < 8; ++i) zero[i] = 0.f;
  v8f acc[4] = { zero, zero, zero, zero };

#pragma unroll
  for (int kk = 0; kk < K; kk += 32) {
    // ISA 16-bit A 16x32 layout: lanes 0-15 hold K=+[0..7],[16..23];
    // lanes 16-31 hold K=+[8..15],[24..31]
    const int kO0 = kk + (hsel ? 8 : 0);
    const int kO1 = kk + (hsel ? 24 : 16);
    const __bf16* ar = (GATHER && kk >= 256) ? arow1 : arow0;
    v8bf alo = *(const v8bf*)(ar + kO0);
    v8bf ahi = *(const v8bf*)(ar + kO1);
    v16bf av = __builtin_shufflevector(alo, ahi,
        0, 1, 2, 3, 4, 5, 6, 7, 8, 9, 10, 11, 12, 13, 14, 15);
#pragma unroll
    for (int t = 0; t < 4; ++t) {
      v8bf blo = *(const v8bf*)(brow[t] + kO0);
      v8bf bhi = *(const v8bf*)(brow[t] + kO1);
      v16bf bv = __builtin_shufflevector(blo, bhi,
          0, 1, 2, 3, 4, 5, 6, 7, 8, 9, 10, 11, 12, 13, 14, 15);
      acc[t] = __builtin_amdgcn_wmma_f32_16x16x32_bf16(
          false, av, false, bv, (short)0, acc[t], false, false);
    }
  }

  // ---- epilogue ----
  if (EPI == 0) {
#pragma unroll
    for (int t = 0; t < 4; ++t) {
      int n = n0 + t * 16 + mr;
      float bn = bias[n];
#pragma unroll
      for (int r = 0; r < 8; ++r) {
        int m = m0 + r + hsel * 8;
        if (m < M) out[(long long)m * Nout + n] = acc[t][r] + bn;
      }
    }
  } else {
    float p0[8], p1[8];
#pragma unroll
    for (int r = 0; r < 8; ++r) { p0[r] = 0.f; p1[r] = 0.f; }
#pragma unroll
    for (int t = 0; t < 4; ++t) {
      int n = n0 + t * 16 + mr;
      float bn = bias[n];
      float wa = w2[n];
      float wb = (EPI == 2) ? w2[256 + n] : 0.f;
#pragma unroll
      for (int r = 0; r < 8; ++r) {
        float v = acc[t][r] + bn;
        v = v > 0.f ? v : 0.f;          // ReLU
        p0[r] += v * wa;
        if (EPI == 2) p1[r] += v * wb;
      }
    }
    for (int mask = 1; mask < 16; mask <<= 1) {   // reduce 16 col-lanes
#pragma unroll
      for (int r = 0; r < 8; ++r) {
        p0[r] += __shfl_xor(p0[r], mask, 32);
        if (EPI == 2) p1[r] += __shfl_xor(p1[r], mask, 32);
      }
    }
    if (mr == 0) {
#pragma unroll
      for (int r = 0; r < 8; ++r) {
        int m = m0 + r + hsel * 8;
        if (m < M) {
          if (EPI == 1) {
            atomicAdd(&out[m], p0[r]);
          } else {
            atomicAdd(&out[(long long)m * 2 + 0], p0[r]);
            atomicAdd(&out[(long long)m * 2 + 1], p1[r]);
          }
        }
      }
    }
  }
}

// ---------------------------------------------------------------------------
// attention kernels
// ---------------------------------------------------------------------------
__global__ void init_attn_kernel(unsigned* __restrict__ amax, float* __restrict__ denom, int n) {
  int i = blockIdx.x * blockDim.x + threadIdx.x;
  if (i < n) { amax[i] = 0u; denom[i] = 0.f; }
}

__global__ void attn_logits_kernel(const float* __restrict__ q, const float* __restrict__ k,
                                   const long long* __restrict__ srcI, const long long* __restrict__ dstI,
                                   float* __restrict__ alpha, unsigned* __restrict__ amax, int E_)
{
  int idx = blockIdx.x * blockDim.x + threadIdx.x;
  if (idx >= E_ * 8) return;
  int e = idx >> 3, hh = idx & 7;
  long long s = srcI[e], d = dstI[e];
  const float4* qp = (const float4*)(q + d * 256 + hh * 32);
  const float4* kp = (const float4*)(k + s * 256 + hh * 32);
  float acc = 0.f;
#pragma unroll
  for (int i = 0; i < 8; ++i) {
    float4 a = qp[i], b = kp[i];
    acc = fmaf(a.x, b.x, fmaf(a.y, b.y, fmaf(a.z, b.z, fmaf(a.w, b.w, acc))));
  }
  acc *= 0.17677669529663687f;   // 1/sqrt(32)
  alpha[idx] = acc;
  atomicMax(amax + d * 8 + hh, f32_enc(acc));
}

__global__ void attn_exp_kernel(float* __restrict__ alpha, const long long* __restrict__ dstI,
                                const unsigned* __restrict__ amax, float* __restrict__ denom, int E_)
{
  int idx = blockIdx.x * blockDim.x + threadIdx.x;
  if (idx >= E_ * 8) return;
  int e = idx >> 3, hh = idx & 7;
  long long d = dstI[e];
  float m = f32_dec(amax[d * 8 + hh]);
  float ex = __expf(alpha[idx] - m);
  alpha[idx] = ex;                       // ea in place
  atomicAdd(denom + d * 8 + hh, ex);
}

__global__ __launch_bounds__(256) void attn_scatter_kernel(
    const float* __restrict__ ea, const float* __restrict__ denom,
    const float* __restrict__ v,
    const long long* __restrict__ srcI, const long long* __restrict__ dstI,
    float* __restrict__ outacc)
{
  int e  = blockIdx.x;        // one block per edge
  int dd = threadIdx.x;       // channel 0..255
  int hh = dd >> 5;
  long long s = srcI[e], d = dstI[e];
  float coef = ea[(long long)e * 8 + hh] / (denom[d * 8 + hh] + 1e-16f);
  atomicAdd(outacc + d * 256 + dd, coef * v[s * 256 + dd]);
}

// h = LayerNorm(h + s)*g + b ; also writes bf16 mirror hbf. One wave32 per row.
__global__ __launch_bounds__(256) void resid_ln_kernel(
    float* __restrict__ h, __bf16* __restrict__ hbf, const float* __restrict__ s,
    const float* __restrict__ g, const float* __restrict__ b, int rows)
{
  int lane = threadIdx.x & 31, wave = threadIdx.x >> 5;
  int row = blockIdx.x * 8 + wave;
  if (row >= rows) return;
  const float4* hp = (const float4*)(h + (long long)row * 256);
  const float4* sp = (const float4*)(s + (long long)row * 256);
  float4 a0 = hp[lane * 2 + 0], a1 = hp[lane * 2 + 1];
  float4 b0 = sp[lane * 2 + 0], b1 = sp[lane * 2 + 1];
  float vals[8] = { a0.x + b0.x, a0.y + b0.y, a0.z + b0.z, a0.w + b0.w,
                    a1.x + b1.x, a1.y + b1.y, a1.z + b1.z, a1.w + b1.w };
  float sum = 0.f, sq = 0.f;
#pragma unroll
  for (int i = 0; i < 8; ++i) { sum += vals[i]; sq += vals[i] * vals[i]; }
  for (int mask = 1; mask < 32; mask <<= 1) {
    sum += __shfl_xor(sum, mask, 32);
    sq  += __shfl_xor(sq,  mask, 32);
  }
  float mu   = sum * (1.f / 256.f);
  float var  = sq * (1.f / 256.f) - mu * mu;
  float rstd = rsqrtf(var + 1e-5f);
  int d0 = lane * 8;
  float o[8];
#pragma unroll
  for (int i = 0; i < 8; ++i) o[i] = (vals[i] - mu) * rstd * g[d0 + i] + b[d0 + i];
  float4* op = (float4*)(h + (long long)row * 256);
  op[lane * 2 + 0] = make_float4(o[0], o[1], o[2], o[3]);
  op[lane * 2 + 1] = make_float4(o[4], o[5], o[6], o[7]);
  uint4 pk;
  pk.x = (unsigned)f2bfb(o[0]) | ((unsigned)f2bfb(o[1]) << 16);
  pk.y = (unsigned)f2bfb(o[2]) | ((unsigned)f2bfb(o[3]) << 16);
  pk.z = (unsigned)f2bfb(o[4]) | ((unsigned)f2bfb(o[5]) << 16);
  pk.w = (unsigned)f2bfb(o[6]) | ((unsigned)f2bfb(o[7]) << 16);
  ((uint4*)(hbf + (long long)row * 256))[lane] = pk;
}

// ---------------------------------------------------------------------------
__global__ void init_out_kernel(float* __restrict__ logits, float* __restrict__ pacc,
                                const float* __restrict__ bec2, const float* __restrict__ bpp2, int E_)
{
  int i = blockIdx.x * blockDim.x + threadIdx.x;
  if (i < 2 * E_) pacc[i] = bpp2[i & 1];
  if (i < E_)     logits[i] = bec2[0];
}

__global__ void softplus_kernel(const float* __restrict__ pacc, float* __restrict__ outp, int n)
{
  int i = blockIdx.x * blockDim.x + threadIdx.x;
  if (i >= n) return;
  float x = pacc[i];
  outp[i] = (x > 0.f) ? (x + log1pf(__expf(-x))) : log1pf(__expf(x));
}

// ---------------------------------------------------------------------------
extern "C" void kernel_launch(void* const* d_in, const int* in_sizes, int n_in,
                              void* d_out, int out_size, void* d_ws, size_t ws_size,
                              hipStream_t stream)
{
  (void)n_in; (void)out_size; (void)ws_size;
  const float*      x    = (const float*)d_in[0];
  const long long*  ei   = (const long long*)d_in[1];   // int64 [2,E]
  const float*      W_in = (const float*)d_in[2];
  const float*      b_in = (const float*)d_in[3];
  const float*      Wq   = (const float*)d_in[4];
  const float*      bq   = (const float*)d_in[5];
  const float*      Wk   = (const float*)d_in[6];
  const float*      bk   = (const float*)d_in[7];
  const float*      Wv   = (const float*)d_in[8];
  const float*      bv   = (const float*)d_in[9];
  const float*      Wsk  = (const float*)d_in[10];
  const float*      bsk  = (const float*)d_in[11];
  const float*      ln_g = (const float*)d_in[12];
  const float*      ln_b = (const float*)d_in[13];
  const float*      Wec1 = (const float*)d_in[14];
  const float*      bec1 = (const float*)d_in[15];
  const float*      Wec2 = (const float*)d_in[16];
  const float*      bec2 = (const float*)d_in[17];
  const float*      Wpp1 = (const float*)d_in[18];
  const float*      bpp1 = (const float*)d_in[19];
  const float*      Wpp2 = (const float*)d_in[20];
  const float*      bpp2 = (const float*)d_in[21];

  const int N = in_sizes[0] / 128;   // 50000
  const int E = in_sizes[1] / 2;     // 400000
  const int D = 256;

  const long long* srcI = ei;
  const long long* dstI = ei + E;

  // f32 workspace: h | bufA | bufB | alpha(E*8) | amax(N*8,u32) | denom(N*8) | pacc(E*2)
  float* h     = (float*)d_ws;
  float* bufA  = h    + (size_t)N * D;
  float* bufB  = bufA + (size_t)N * D;
  float* alpha = bufB + (size_t)N * D;
  unsigned* amax = (unsigned*)(alpha + (size_t)E * 8);
  float* denom = (float*)(amax + (size_t)N * 8);
  float* pacc  = denom + (size_t)N * 8;
  // bf16 arenas
  __bf16* xbf = (__bf16*)(pacc + (size_t)E * 2);
  __bf16* hbf = xbf + (size_t)N * 128;
  __bf16* wbf = hbf + (size_t)N * 256;
  // weight arena offsets (elements)
  const size_t o_Win = 0, o_Wq = 32768, o_Wk = 229376, o_Wv = 425984,
               o_Ws = 622592, o_Wec1 = 819200, o_Wpp1 = 950272;
  float* outF = (float*)d_out;

  dim3 blk(256, 1, 1);
  dim3 gN(D / 64, (N + 127) / 128, 1);
  dim3 gE(D / 64, (E + 127) / 128, 1);
  const int nh = N * 8, eh = E * 8;
  auto cvtg = [](int n8) { return dim3((n8 + 255) / 256, 1, 1); };

  // ---- pre-convert weights + x to bf16 arenas ----
  cvt_bf16_kernel<<<cvtg(32768 / 8), blk, 0, stream>>>(W_in, wbf + o_Win, 32768 / 8);
  cvt_bf16_kernel<<<cvtg(196608 / 8), blk, 0, stream>>>(Wq,  wbf + o_Wq,  196608 / 8);
  cvt_bf16_kernel<<<cvtg(196608 / 8), blk, 0, stream>>>(Wk,  wbf + o_Wk,  196608 / 8);
  cvt_bf16_kernel<<<cvtg(196608 / 8), blk, 0, stream>>>(Wv,  wbf + o_Wv,  196608 / 8);
  cvt_bf16_kernel<<<cvtg(196608 / 8), blk, 0, stream>>>(Wsk, wbf + o_Ws,  196608 / 8);
  cvt_bf16_kernel<<<cvtg(131072 / 8), blk, 0, stream>>>(Wec1, wbf + o_Wec1, 131072 / 8);
  cvt_bf16_kernel<<<cvtg(131072 / 8), blk, 0, stream>>>(Wpp1, wbf + o_Wpp1, 131072 / 8);
  cvt_bf16_kernel<<<cvtg(N * 128 / 8), blk, 0, stream>>>(x, xbf, N * 128 / 8);

  // ---- h = x @ W_in^T + b_in ; bf16 mirror ----
  gemm_wmma_kernel<128, 0, 0><<<gN, blk, 0, stream>>>(
      xbf, nullptr, nullptr, nullptr, wbf + o_Win, b_in, nullptr, h, N, D);
  cvt_bf16_kernel<<<cvtg(N * 256 / 8), blk, 0, stream>>>(h, hbf, N * 256 / 8);

  for (int l = 0; l < 3; ++l) {
    const size_t wo = (size_t)l * D * D, bo = (size_t)l * D;
    gemm_wmma_kernel<256, 0, 0><<<gN, blk, 0, stream>>>(          // q -> bufA
        hbf, nullptr, nullptr, nullptr, wbf + o_Wq + wo, bq + bo, nullptr, bufA, N, D);
    gemm_wmma_kernel<256, 0, 0><<<gN, blk, 0, stream>>>(          // k -> bufB
        hbf, nullptr, nullptr, nullptr, wbf + o_Wk + wo, bk + bo, nullptr, bufB, N, D);
    init_attn_kernel<<<(nh + 255) / 256, blk, 0, stream>>>(amax, denom, nh);
    attn_logits_kernel<<<(eh + 255) / 256, blk, 0, stream>>>(bufA, bufB, srcI, dstI, alpha, amax, E);
    attn_exp_kernel<<<(eh + 255) / 256, blk, 0, stream>>>(alpha, dstI, amax, denom, E);
    gemm_wmma_kernel<256, 0, 0><<<gN, blk, 0, stream>>>(          // v -> bufA
        hbf, nullptr, nullptr, nullptr, wbf + o_Wv + wo, bv + bo, nullptr, bufA, N, D);
    gemm_wmma_kernel<256, 0, 0><<<gN, blk, 0, stream>>>(          // skip -> bufB
        hbf, nullptr, nullptr, nullptr, wbf + o_Ws + wo, bsk + bo, nullptr, bufB, N, D);
    attn_scatter_kernel<<<E, blk, 0, stream>>>(alpha, denom, bufA, srcI, dstI, bufB);
    resid_ln_kernel<<<(N + 7) / 8, blk, 0, stream>>>(h, hbf, bufB, ln_g + bo, ln_b + bo, N);
  }

  // ---- edge heads: fused gather-GEMM + ReLU + second-linear epilogue ----
  init_out_kernel<<<(2 * E + 255) / 256, blk, 0, stream>>>(outF, pacc, bec2, bpp2, E);
  gemm_wmma_kernel<512, 1, 1><<<gE, blk, 0, stream>>>(
      nullptr, hbf, srcI, dstI, wbf + o_Wec1, bec1, Wec2, outF, E, D);
  gemm_wmma_kernel<512, 1, 2><<<gE, blk, 0, stream>>>(
      nullptr, hbf, srcI, dstI, wbf + o_Wpp1, bpp1, Wpp2, pacc, E, D);
  softplus_kernel<<<(2 * E + 255) / 256, blk, 0, stream>>>(pacc, outF + E, 2 * E);
}
